// GCN_77008763617734
// MI455X (gfx1250) — matrix operand order repaired and checked by reading, not compiled, not run on hardware
//
#include <hip/hip_runtime.h>

// ---------------------------------------------------------------------------
// GCN forward on MI455X (gfx1250): bf16 WMMA GEMMs with f32 accumulate.
//   h0 = x @ W1                 (8192x512)    bf16 out
//   h1 = relu(adj @ h0 + b1)    (8192x512)    f32 out
//   h2 = h1 @ W2pad             (8192x384)    bf16 out (cols 345..383 are 0)
//   out= adj @ h2 + b2pad       (8192x345)    f32 out (store-guarded at 345)
//
// Grid: blockIdx.x = N-block (fastest varying), blockIdx.y = M-block, so the
// N-blocks sharing one adj block-row dispatch adjacently and reuse adj in L2
// (adj = 256MB > 192MB L2; this keeps adj HBM traffic at ~1x per GEMM).
// ---------------------------------------------------------------------------

typedef __attribute__((ext_vector_type(16))) __bf16 v16bf;
typedef __attribute__((ext_vector_type(8)))  float  v8f;
typedef __attribute__((ext_vector_type(4)))  float  vf4;
typedef __attribute__((ext_vector_type(16))) float  v16f;

union V16B { v16bf v; unsigned u[8]; };

// A fragment: 16x32 bf16, this lane holds row m, K = c0..c0+7 and c0+16..c0+23
// (c0 = k + (lane>=16 ? 8 : 0)), packed K-pairs per dword -> 8 VGPRs.
// Vector convert -> clang emits native v_cvt_pk_bf16_f32.
__device__ __forceinline__ v16bf load_a_frag_f32(const float* __restrict__ p) {
    vf4 f0 = *(const vf4*)(p);
    vf4 f1 = *(const vf4*)(p + 4);
    vf4 f2 = *(const vf4*)(p + 16);
    vf4 f3 = *(const vf4*)(p + 20);
    v8f ab = __builtin_shufflevector(f0, f1, 0, 1, 2, 3, 4, 5, 6, 7);
    v8f cd = __builtin_shufflevector(f2, f3, 0, 1, 2, 3, 4, 5, 6, 7);
    v16f all = __builtin_shufflevector(ab, cd, 0, 1, 2, 3, 4, 5, 6, 7,
                                               8, 9, 10, 11, 12, 13, 14, 15);
    return __builtin_convertvector(all, v16bf);
}

// B fragment: 32x16 bf16, lane holds column n = lane&15, K = c0..c0+7 and c0+16..c0+23 (c0 = k + 8*half).
__device__ __forceinline__ v16bf load_b_frag(const __bf16* __restrict__ p, int ldb) {
    v16bf r;
#pragma unroll
    for (int s = 0; s < 8; ++s) {
        r[s]     = p[(size_t)s * ldb];
        r[s + 8] = p[(size_t)(s + 16) * ldb];
    }
    return r;
}

// ---------------------------------------------------------------------------
// Generic GEMM: C[MxN] = op(A[MxK]_f32 x B[KxN]_bf16 + bias)
// Block = 256 threads = 8 waves arranged 4(M) x 2(N); wave tile 32x64;
// block tile 128x128. Requires M%128==0, N%128==0 (padded), K%32==0.
// blockIdx.x = N-block, blockIdx.y = M-block.
// ---------------------------------------------------------------------------
template<bool RELU, bool BIAS, bool OUT_BF16>
__global__ __launch_bounds__(256)
void gcn_gemm_wmma(const float* __restrict__ A, const __bf16* __restrict__ B,
                   const float* __restrict__ bias, void* __restrict__ Cv,
                   int K, int lda, int ldb, int ldc, int n_max)
{
    const int lane = threadIdx.x & 31;
    const int wave = threadIdx.x >> 5;
    const int wm   = wave >> 1;          // 0..3
    const int wn   = wave & 1;           // 0..1
    const int half = lane >> 4;          // 0/1
    const int r    = lane & 15;

    const long m_blk  = (long)blockIdx.y * 128 + wm * 32;
    const int  n_base = blockIdx.x * 128 + wn * 64;

    v8f acc[2][4];
#pragma unroll
    for (int i = 0; i < 2; ++i)
#pragma unroll
        for (int j = 0; j < 4; ++j)
            acc[i][j] = (v8f){0.f,0.f,0.f,0.f,0.f,0.f,0.f,0.f};

    const float*  Arow0 = A + (m_blk + r) * (long)lda + half * 8;
    const float*  Arow1 = Arow0 + 16l * lda;
    const __bf16* Brow  = B + (size_t)(half * 8) * ldb + n_base + r;

    for (int k = 0; k < K; k += 32) {
        // prefetch adj a few K-steps ahead (speculative; silently dropped OOB)
        __builtin_prefetch(Arow0 + k + 256, 0, 1);

        v16bf a0 = load_a_frag_f32(Arow0 + k);
        v16bf a1 = load_a_frag_f32(Arow1 + k);
        const __bf16* bp = Brow + (size_t)k * ldb;
        v16bf b0 = load_b_frag(bp, ldb);
        v16bf b1 = load_b_frag(bp + 16, ldb);
        v16bf b2 = load_b_frag(bp + 32, ldb);
        v16bf b3 = load_b_frag(bp + 48, ldb);

        acc[0][0] = __builtin_amdgcn_wmma_f32_16x16x32_bf16(false, a0, false, b0, (short)0, acc[0][0], false, false);
        acc[0][1] = __builtin_amdgcn_wmma_f32_16x16x32_bf16(false, a0, false, b1, (short)0, acc[0][1], false, false);
        acc[0][2] = __builtin_amdgcn_wmma_f32_16x16x32_bf16(false, a0, false, b2, (short)0, acc[0][2], false, false);
        acc[0][3] = __builtin_amdgcn_wmma_f32_16x16x32_bf16(false, a0, false, b3, (short)0, acc[0][3], false, false);
        acc[1][0] = __builtin_amdgcn_wmma_f32_16x16x32_bf16(false, a1, false, b0, (short)0, acc[1][0], false, false);
        acc[1][1] = __builtin_amdgcn_wmma_f32_16x16x32_bf16(false, a1, false, b1, (short)0, acc[1][1], false, false);
        acc[1][2] = __builtin_amdgcn_wmma_f32_16x16x32_bf16(false, a1, false, b2, (short)0, acc[1][2], false, false);
        acc[1][3] = __builtin_amdgcn_wmma_f32_16x16x32_bf16(false, a1, false, b3, (short)0, acc[1][3], false, false);
    }

    // Epilogue. C/D layout: VGPR i -> (M = i + 8*half, N = lane&15) per 16x16 tile.
#pragma unroll
    for (int mt = 0; mt < 2; ++mt) {
#pragma unroll
        for (int nt = 0; nt < 4; ++nt) {
            const int n = n_base + nt * 16 + r;
            float bv = 0.f;
            if (BIAS) bv = bias[n];           // bias arrays are padded to N
            if (n < n_max) {
                const long mrow = m_blk + mt * 16 + half * 8;
#pragma unroll
                for (int i = 0; i < 8; ++i) {
                    float v = acc[mt][nt][i] + bv;
                    if (RELU) v = fmaxf(v, 0.f);
                    const long off = (mrow + i) * (long)ldc + n;
                    if (OUT_BF16) ((__bf16*)Cv)[off] = (__bf16)v;
                    else          ((float*)Cv)[off]  = v;
                }
            }
        }
    }
}

// ---------------------------------------------------------------------------
// Converters
// ---------------------------------------------------------------------------
__global__ void cvt_f32_to_bf16(const float* __restrict__ in, __bf16* __restrict__ out, int n) {
    int i = blockIdx.x * blockDim.x + threadIdx.x;
    if (i < n) out[i] = (__bf16)in[i];
}

// W2: 512x345 f32 -> 512x384 bf16 (zero-padded columns)
__global__ void cvt_pad_w2(const float* __restrict__ in, __bf16* __restrict__ out) {
    int i = blockIdx.x * blockDim.x + threadIdx.x;
    if (i < 512 * 384) {
        int row = i / 384, col = i % 384;
        out[i] = (col < 345) ? (__bf16)in[row * 345 + col] : (__bf16)0.0f;
    }
}

// b2: 345 f32 -> 384 f32 (zero-padded)
__global__ void cvt_pad_b2(const float* __restrict__ in, float* __restrict__ out) {
    int i = blockIdx.x * blockDim.x + threadIdx.x;
    if (i < 384) out[i] = (i < 345) ? in[i] : 0.0f;
}

// ---------------------------------------------------------------------------
// Launch
// ---------------------------------------------------------------------------
extern "C" void kernel_launch(void* const* d_in, const int* in_sizes, int n_in,
                              void* d_out, int out_size, void* d_ws, size_t ws_size,
                              hipStream_t stream) {
    constexpr int N_NODES = 8192, F_IN = 512, NCP = 384, NC = 345;

    const float* x   = (const float*)d_in[0];
    const float* adj = (const float*)d_in[1];
    const float* W1  = (const float*)d_in[2];
    const float* b1  = (const float*)d_in[3];
    const float* W2  = (const float*)d_in[4];
    const float* b2  = (const float*)d_in[5];

    char* ws = (char*)d_ws;
    __bf16* W1bf = (__bf16*)(ws);                       //   512*512*2 =   524288
    __bf16* W2bf = (__bf16*)(ws +   524288);            //   512*384*2 =   393216
    float*  b2p  = (float*) (ws +   917504);            //   384*4     =     1536
    __bf16* h0   = (__bf16*)(ws +   919040);            //  8192*512*2 =  8388608
    float*  h1   = (float*) (ws +  9307648);            //  8192*512*4 = 16777216
    __bf16* h2   = (__bf16*)(ws + 26084864);            //  8192*384*2 =  6291456
    // total workspace: 32376320 bytes (~30.9 MB)

    // --- converters ---
    cvt_f32_to_bf16<<<(F_IN * F_IN + 255) / 256, 256, 0, stream>>>(W1, W1bf, F_IN * F_IN);
    cvt_pad_w2<<<(F_IN * NCP + 255) / 256, 256, 0, stream>>>(W2, W2bf);
    cvt_pad_b2<<<2, 256, 0, stream>>>(b2, b2p);

    dim3 blk(256);

    // h0 = x @ W1                      M=8192 N=512 K=512
    gcn_gemm_wmma<false, false, true><<<dim3(F_IN / 128, N_NODES / 128), blk, 0, stream>>>(
        x, W1bf, nullptr, (void*)h0, F_IN, F_IN, F_IN, F_IN, F_IN);

    // h1 = relu(adj @ h0 + b1)         M=8192 N=512 K=8192
    gcn_gemm_wmma<true, true, false><<<dim3(F_IN / 128, N_NODES / 128), blk, 0, stream>>>(
        adj, h0, b1, (void*)h1, N_NODES, N_NODES, F_IN, F_IN, F_IN);

    // h2 = h1 @ W2pad                  M=8192 N=384 K=512
    gcn_gemm_wmma<false, false, true><<<dim3(NCP / 128, N_NODES / 128), blk, 0, stream>>>(
        h1, W2bf, nullptr, (void*)h2, F_IN, F_IN, NCP, NCP, NCP);

    // out = adj @ h2 + b2pad           M=8192 N=384(pad) K=8192, store n<345
    gcn_gemm_wmma<false, true, false><<<dim3(NCP / 128, N_NODES / 128), blk, 0, stream>>>(
        adj, h2, b2p, d_out, N_NODES, N_NODES, NCP, NC, NC);
}
